// Pointnet2MSG_28690381537640
// MI455X (gfx1250) — matrix operand order repaired and checked by reading, not compiled, not run on hardware
//
#include <hip/hip_runtime.h>

// ---------------------------------------------------------------------------
// PointNet++ MSG forward for MI455X (gfx1250, wave32, WMMA).
//  * MLP GEMMs: v_wmma_f32_16x16x32_f16, one wave per 16x16 C tile,
//    k-loop unrolled 2x with independent register sets (clause of 8 b128
//    loads, partial s_wait_loadcnt overlap; no register rotation moves),
//    fused bias+ReLU epilogue; final SA layer fuses the max-pool over K
//    (per-lane row max + shfl_xor(16) across half-waves; K=32 uses 2 M-tiles
//    per wave sharing one B fragment).
//  * Whole working set (~60MB) is L2-resident (192MB), so direct global
//    fragment loads run at L2 bandwidth; waves in a block share the same
//    weight panel for WGP$ hits.
//  * FPS: one workgroup/batch, dist array in LDS, two-level wave32
//    shfl_xor argmax (3 barriers / iteration).
// ---------------------------------------------------------------------------

typedef _Float16 v8h  __attribute__((ext_vector_type(8)));
typedef _Float16 v16h __attribute__((ext_vector_type(16)));
typedef float    v8f  __attribute__((ext_vector_type(8)));

__device__ __forceinline__ v16h cat16(v8h a, v8h b) {
  return __builtin_shufflevector(a, b, 0,1,2,3,4,5,6,7,8,9,10,11,12,13,14,15);
}

// ------------------------------ split input --------------------------------
__global__ void split_pc_kernel(const float* __restrict__ pc,
                                float* __restrict__ xyz, float* __restrict__ feat) {
  int i = blockIdx.x * blockDim.x + threadIdx.x;      // over B*N
  if (i >= 4 * 8192) return;
  const float* p = pc + (size_t)i * 9;
  xyz[i * 3 + 0] = p[0]; xyz[i * 3 + 1] = p[1]; xyz[i * 3 + 2] = p[2];
  #pragma unroll
  for (int c = 0; c < 6; ++c) feat[(size_t)i * 6 + c] = p[3 + c];
}

// ------------------------------ weight pack --------------------------------
// Wt layout: [NP][KP] f16 (transposed, zero padded) so each lane's B fragment
// is two contiguous runs of 8 f16 per k-step.
__global__ void pack_w_kernel(const float* __restrict__ W, _Float16* __restrict__ Wt,
                              int Cin, int Cout, int KP, int NP) {
  int e = blockIdx.x * blockDim.x + threadIdx.x;
  if (e >= KP * NP) return;
  int n = e / KP, k = e % KP;
  float v = (k < Cin && n < Cout) ? W[(size_t)k * Cout + n] : 0.0f;
  Wt[e] = (_Float16)v;
}

// --------------------------------- FPS -------------------------------------
__global__ __launch_bounds__(1024) void fps_kernel(const float* __restrict__ xyz,
                                                   int N, int npoint,
                                                   int* __restrict__ out_idx) {
  __shared__ float dist[8192];
  __shared__ float swd[32];
  __shared__ int   swi[32];
  __shared__ float cpt[3];
  __shared__ int   sfar;
  const int b = blockIdx.x, t = threadIdx.x;
  const float* bx = xyz + (size_t)b * N * 3;
  for (int i = t; i < N; i += 1024) dist[i] = 1e10f;
  if (t == 0) sfar = 0;
  __syncthreads();
  for (int it = 0; it < npoint; ++it) {
    if (t == 0) {
      int f = sfar;
      out_idx[b * npoint + it] = f;
      cpt[0] = bx[f * 3]; cpt[1] = bx[f * 3 + 1]; cpt[2] = bx[f * 3 + 2];
    }
    __syncthreads();
    float cx = cpt[0], cy = cpt[1], cz = cpt[2];
    float bd = -1.0f; int bi = 0;
    for (int i = t; i < N; i += 1024) {
      float dx = bx[i * 3] - cx, dy = bx[i * 3 + 1] - cy, dz = bx[i * 3 + 2] - cz;
      float d = fminf(dist[i], dx * dx + dy * dy + dz * dz);
      dist[i] = d;
      if (d > bd) { bd = d; bi = i; }          // strict > keeps lowest index
    }
    #pragma unroll
    for (int o = 16; o > 0; o >>= 1) {         // wave32 argmax, min-index ties
      float od = __shfl_xor(bd, o, 32);
      int   oi = __shfl_xor(bi, o, 32);
      if (od > bd || (od == bd && oi < bi)) { bd = od; bi = oi; }
    }
    if ((t & 31) == 0) { swd[t >> 5] = bd; swi[t >> 5] = bi; }
    __syncthreads();
    if (t < 32) {
      bd = swd[t]; bi = swi[t];
      #pragma unroll
      for (int o = 16; o > 0; o >>= 1) {
        float od = __shfl_xor(bd, o, 32);
        int   oi = __shfl_xor(bi, o, 32);
        if (od > bd || (od == bd && oi < bi)) { bd = od; bi = oi; }
      }
      if (t == 0) sfar = bi;
    }
    __syncthreads();
  }
}

// ------------------------------ gather xyz ---------------------------------
__global__ void gather_xyz_kernel(const float* __restrict__ src, const int* __restrict__ idx,
                                  float* __restrict__ dst, int M, int Nsrc) {
  int i = blockIdx.x * blockDim.x + threadIdx.x;   // over B*M
  if (i >= 4 * M) return;
  int b = i / M;
  int id = idx[i];
  #pragma unroll
  for (int c = 0; c < 3; ++c)
    dst[(size_t)i * 3 + c] = src[((size_t)b * Nsrc + id) * 3 + c];
}

// ------------------------------ ball query ---------------------------------
// First K points (in index order) with d2 < r2; pad with first hit (or 0).
__global__ void ball_query_kernel(const float* __restrict__ nxyz, const float* __restrict__ xyz,
                                  int M, int N, float r2, int K, int* __restrict__ out) {
  int wid  = (int)((blockIdx.x * blockDim.x + threadIdx.x) >> 5);
  int lane = threadIdx.x & 31;
  if (wid >= 4 * M) return;
  int b = wid / M;
  const float* q = nxyz + (size_t)wid * 3;
  float qx = q[0], qy = q[1], qz = q[2];
  const float* px = xyz + (size_t)b * N * 3;
  int* o = out + (size_t)wid * K;
  int count = 0, first = 0;
  bool haveFirst = false;
  for (int j0 = 0; j0 < N && count < K; j0 += 32) {
    int j = j0 + lane;
    bool hit = false;
    if (j < N) {
      float dx = px[j * 3] - qx, dy = px[j * 3 + 1] - qy, dz = px[j * 3 + 2] - qz;
      hit = (dx * dx + dy * dy + dz * dz) < r2;
    }
    unsigned mask = (unsigned)__ballot(hit);
    if (hit) {
      int pos = count + __popc(mask & ((1u << lane) - 1u));
      if (pos < K) o[pos] = j;
    }
    if (!haveFirst && mask) { first = j0 + (__ffs(mask) - 1); haveFirst = true; }
    count += __popc(mask);
  }
  if (count > K) count = K;
  for (int p = count + lane; p < K; p += 32) o[p] = first;
}

// ---------------------- SA grouped-feature build (f16) ---------------------
__global__ void build_sa_x_kernel(const float* __restrict__ xyz, const float* __restrict__ feats,
                                  const float* __restrict__ nxyz, const int* __restrict__ gi,
                                  _Float16* __restrict__ Xb, int M, int N, int K, int Cf,
                                  int P, int rows, int rowsP) {
  long long e = (long long)blockIdx.x * blockDim.x + threadIdx.x;
  long long tot = (long long)rowsP * P;
  if (e >= tot) return;
  int row = (int)(e / P), c = (int)(e % P);
  float v = 0.0f;
  if (row < rows) {
    int g = row / K;             // b*M + m
    int b = g / M;
    if (c < 3) {
      int id = gi[row];
      v = xyz[((size_t)b * N + id) * 3 + c] - nxyz[(size_t)g * 3 + c];
    } else if (c < 3 + Cf) {
      int id = gi[row];
      v = feats[((size_t)b * N + id) * Cf + (c - 3)];
    }
  }
  Xb[e] = (_Float16)v;
}

// ------------------------------- three-NN ----------------------------------
__global__ void three_nn_kernel(const float* __restrict__ uxyz, const float* __restrict__ kxyz,
                                int Nu, int Nk, int* __restrict__ nnidx, float* __restrict__ nnw) {
  int i = blockIdx.x * blockDim.x + threadIdx.x;   // over B*Nu
  if (i >= 4 * Nu) return;
  int b = i / Nu;
  float qx = uxyz[i * 3], qy = uxyz[i * 3 + 1], qz = uxyz[i * 3 + 2];
  const float* kb = kxyz + (size_t)b * Nk * 3;
  float d0 = 3.4e38f, d1 = 3.4e38f, d2v = 3.4e38f;
  int i0 = 0, i1 = 0, i2 = 0;
  for (int j = 0; j < Nk; ++j) {
    float dx = kb[j * 3] - qx, dy = kb[j * 3 + 1] - qy, dz = kb[j * 3 + 2] - qz;
    float d = dx * dx + dy * dy + dz * dz;
    if (d < d0)      { d2v = d1; i2 = i1; d1 = d0; i1 = i0; d0 = d; i0 = j; }
    else if (d < d1) { d2v = d1; i2 = i1; d1 = d;  i1 = j; }
    else if (d < d2v){ d2v = d;  i2 = j; }
  }
  float w0 = 1.0f / (sqrtf(fmaxf(d0, 0.0f)) + 1e-8f);
  float w1 = 1.0f / (sqrtf(fmaxf(d1, 0.0f)) + 1e-8f);
  float w2 = 1.0f / (sqrtf(fmaxf(d2v, 0.0f)) + 1e-8f);
  float s = w0 + w1 + w2;
  nnidx[(size_t)i * 3 + 0] = i0; nnidx[(size_t)i * 3 + 1] = i1; nnidx[(size_t)i * 3 + 2] = i2;
  nnw[(size_t)i * 3 + 0] = w0 / s; nnw[(size_t)i * 3 + 1] = w1 / s; nnw[(size_t)i * 3 + 2] = w2 / s;
}

// ---------------------- FP concat(interp, u_feats) build -------------------
__global__ void build_fp_x_kernel(const float* __restrict__ kfeats, const float* __restrict__ ufeats,
                                  const int* __restrict__ nnidx, const float* __restrict__ nnw,
                                  _Float16* __restrict__ Xb, int Nu, int Nk, int Ck, int ldk,
                                  int Cu, int ldu, int P, int rows, int rowsP) {
  long long e = (long long)blockIdx.x * blockDim.x + threadIdx.x;
  long long tot = (long long)rowsP * P;
  if (e >= tot) return;
  int row = (int)(e / P), c = (int)(e % P);
  float v = 0.0f;
  if (row < rows) {
    int b = row / Nu;
    if (c < Ck) {
      const int*   id = nnidx + (size_t)row * 3;
      const float* w  = nnw   + (size_t)row * 3;
      const float* kb = kfeats + (size_t)b * Nk * ldk;
      v = w[0] * kb[(size_t)id[0] * ldk + c]
        + w[1] * kb[(size_t)id[1] * ldk + c]
        + w[2] * kb[(size_t)id[2] * ldk + c];
    } else if (c < Ck + Cu) {
      v = ufeats[(size_t)row * ldu + (c - Ck)];
    }
  }
  Xb[e] = (_Float16)v;
}

// -------------------- WMMA GEMM + bias + ReLU (+maxpool) -------------------
// X : [MrowsP, KP] f16, Wt : [NP, KP] f16 (transposed), KP%32==0, NP%16==0.
// MT = M-tiles per wave (2 only for the pooled K=32 path, shares B fragment).
// EPI: 0 = f16 padded out, 1 = f32 out [rows,ldYf], 2 = f32 transposed
//      [B,Cout,8192] (final layer -> d_out), 3 = fused max-pool over 16*MT
//      rows -> f32 feature buffer.
template <int MT, int EPI>
__global__ __launch_bounds__(256) void gemm_mlp_kernel(
    const _Float16* __restrict__ X, const _Float16* __restrict__ Wt,
    const float* __restrict__ bias,
    _Float16* __restrict__ Yh, float* __restrict__ Yf, float* __restrict__ Ypool,
    int Mrows, int MrowsP, int KP, int NP, int Cout,
    int ldYf, int poolStride, int poolOfs) {
  const int wave = (int)((blockIdx.x * blockDim.x + threadIdx.x) >> 5);
  const int lane = threadIdx.x & 31;
  const int tgCount = (MrowsP >> 4) / MT;
  if (wave >= (NP >> 4) * tgCount) return;    // wave-uniform exit (EXEC all-1s)
  const int tg = wave % tgCount;              // fast index: block shares B panel
  const int tn = wave / tgCount;
  const int l15 = lane & 15;
  const int hs  = lane >> 4;
  const int col = (tn << 4) + l15;

  // B fragment: lane holds column `col`; k = kk + hs*8 + j (j<8) / kk+16+hs*8 (j>=8)
  const _Float16* wp  = Wt + (size_t)col * KP + hs * 8;
  // A fragment: lane holds row; ch = kk + hs*8 + j (j<8) / kk+16+hs*8 (j>=8)
  const int row0 = tg * MT * 16 + l15;
  const _Float16* xp0 = X + (size_t)row0 * KP + hs * 8;
  const _Float16* xp1 = xp0 + (size_t)16 * KP;

  v8f acc0 = {}; v8f acc1 = {};
  int kk = 0;
  // 2x unrolled main loop: independent register sets per half-step lets the
  // compiler clause all loads and overlap the 2nd half's loads with wmma #1.
  for (; kk + 64 <= KP; kk += 64) {
    v8h b0 = *(const v8h*)(wp + kk);
    v8h b1 = *(const v8h*)(wp + kk + 16);
    v8h b2 = *(const v8h*)(wp + kk + 32);
    v8h b3 = *(const v8h*)(wp + kk + 48);
    v8h a0 = *(const v8h*)(xp0 + kk);
    v8h a1 = *(const v8h*)(xp0 + kk + 16);
    v8h a2 = *(const v8h*)(xp0 + kk + 32);
    v8h a3 = *(const v8h*)(xp0 + kk + 48);
    acc0 = __builtin_amdgcn_wmma_f32_16x16x32_f16(false, cat16(a0, a1), false,
                                                  cat16(b0, b1), (short)0,
                                                  acc0, false, false);
    if (MT == 2) {
      v8h c0 = *(const v8h*)(xp1 + kk);
      v8h c1 = *(const v8h*)(xp1 + kk + 16);
      acc1 = __builtin_amdgcn_wmma_f32_16x16x32_f16(false, cat16(c0, c1), false,
                                                    cat16(b0, b1), (short)0,
                                                    acc1, false, false);
    }
    acc0 = __builtin_amdgcn_wmma_f32_16x16x32_f16(false, cat16(a2, a3), false,
                                                  cat16(b2, b3), (short)0,
                                                  acc0, false, false);
    if (MT == 2) {
      v8h c2 = *(const v8h*)(xp1 + kk + 32);
      v8h c3 = *(const v8h*)(xp1 + kk + 48);
      acc1 = __builtin_amdgcn_wmma_f32_16x16x32_f16(false, cat16(c2, c3), false,
                                                    cat16(b2, b3), (short)0,
                                                    acc1, false, false);
    }
  }
  if (kk < KP) {                               // single 32-wide remainder
    v8h b0 = *(const v8h*)(wp + kk);
    v8h b1 = *(const v8h*)(wp + kk + 16);
    v8h a0 = *(const v8h*)(xp0 + kk);
    v8h a1 = *(const v8h*)(xp0 + kk + 16);
    acc0 = __builtin_amdgcn_wmma_f32_16x16x32_f16(false, cat16(a0, a1), false,
                                                  cat16(b0, b1), (short)0,
                                                  acc0, false, false);
    if (MT == 2) {
      v8h c0 = *(const v8h*)(xp1 + kk);
      v8h c1 = *(const v8h*)(xp1 + kk + 16);
      acc1 = __builtin_amdgcn_wmma_f32_16x16x32_f16(false, cat16(c0, c1), false,
                                                    cat16(b0, b1), (short)0,
                                                    acc1, false, false);
    }
  }

  float bval = (col < Cout) ? bias[col] : 0.0f;

  if (EPI == 3) {                                 // fused max over 16*MT rows
    float r = acc0[0];
    #pragma unroll
    for (int i = 1; i < 8; ++i) r = fmaxf(r, acc0[i]);
    if (MT == 2) {
      #pragma unroll
      for (int i = 0; i < 8; ++i) r = fmaxf(r, acc1[i]);
    }
    r = fmaxf(r, __shfl_xor(r, 16, 32));          // other 8 rows of each tile
    float y = fmaxf(r + bval, 0.0f);
    if (hs == 0 && col < Cout)
      Ypool[(size_t)tg * poolStride + poolOfs + col] = y;
    return;
  }

  #pragma unroll
  for (int i = 0; i < 8; ++i) {                   // D row = hs*8 + i
    int row = tg * MT * 16 + hs * 8 + i;
    float y = fmaxf(acc0[i] + bval, 0.0f);
    if (EPI == 0) Yh[(size_t)row * NP + col] = (_Float16)y;
    if (EPI == 1 && row < Mrows && col < Cout) Yf[(size_t)row * ldYf + col] = y;
    if (EPI == 2 && row < Mrows && col < Cout) { // [B, Cout, 8192] transposed
      int b = row >> 13, n = row & 8191;
      Yf[((size_t)b * Cout + col) * 8192 + n] = y;
    }
  }
}

// ---------------------------------------------------------------------------
static inline int  P32h(int x) { return (x + 31) & ~31; }
static inline int  cdivh(long long a, long long b) { return (int)((a + b - 1) / b); }

extern "C" void kernel_launch(void* const* d_in, const int* in_sizes, int n_in,
                              void* d_out, int out_size, void* d_ws, size_t ws_size,
                              hipStream_t stream) {
  (void)in_sizes; (void)n_in; (void)out_size; (void)ws_size;

  static const int Nl[5]      = {8192, 1000, 500, 250, 125};
  static const int featCap[5] = {6, 256, 512, 512, 1024};
  static const int saK[2]     = {16, 32};
  static const float saR[4][2] = {{0.0175f, 0.025f}, {0.025f, 0.05f},
                                  {0.05f, 0.1f}, {0.1f, 0.2f}};
  static const int saDims[4][2][4] = {
    {{9, 16, 16, 32},     {9, 32, 32, 64}},
    {{99, 64, 64, 128},   {99, 64, 96, 128}},
    {{259, 128, 196, 256},{259, 128, 196, 256}},
    {{515, 256, 256, 512},{515, 256, 384, 512}}};
  static const int fpDims[4][3] = {{262, 128, 128}, {608, 256, 256},
                                   {768, 512, 512}, {1536, 512, 512}};

  const float* pc = (const float*)d_in[0];

  // ---- workspace bump allocator (256B aligned) ----
  char* base = (char*)d_ws;
  size_t off = 0;
  auto walloc = [&](size_t bytes) -> void* {
    void* p = base + off;
    off = (off + bytes + 255) & ~(size_t)255;
    return p;
  };

  float* xyzL[5]; float* featL[5];
  for (int l = 0; l < 5; ++l) xyzL[l]  = (float*)walloc((size_t)4 * Nl[l] * 3 * 4);
  for (int l = 0; l < 5; ++l) featL[l] = (float*)walloc((size_t)4 * Nl[l] * featCap[l] * 4);
  int*   fpsIdx  = (int*)walloc((size_t)4 * 1000 * 4);
  int*   ballIdx = (int*)walloc((size_t)4 * 1000 * 32 * 4);
  int*   nnIdx   = (int*)walloc((size_t)4 * 8192 * 3 * 4);
  float* nnW     = (float*)walloc((size_t)4 * 8192 * 3 * 4);
  _Float16* Xb = (_Float16*)walloc((size_t)10000000 * 2);
  _Float16* Y0 = (_Float16*)walloc((size_t)4500000 * 2);
  _Float16* Y1 = (_Float16*)walloc((size_t)7500000 * 2);

  // ---- weight/bias tables (d_in flatten order: pointcloud, sa..., fp...) ----
  const float* Wp[32]; const float* Bp[32];
  int cinL[32], coutL[32];
  _Float16* WtL[32];
  for (int st = 0; st < 4; ++st)
    for (int sc = 0; sc < 2; ++sc)
      for (int l = 0; l < 3; ++l) {
        int lid = (st * 2 + sc) * 3 + l;
        int din = 1 + 2 * (((st * 2 + sc) * 3) + l);
        Wp[lid] = (const float*)d_in[din];
        Bp[lid] = (const float*)d_in[din + 1];
        cinL[lid] = saDims[st][sc][l];
        coutL[lid] = saDims[st][sc][l + 1];
      }
  for (int i = 0; i < 4; ++i)
    for (int l = 0; l < 2; ++l) {
      int lid = 24 + i * 2 + l;
      int din = 49 + 2 * (i * 2 + l);
      Wp[lid] = (const float*)d_in[din];
      Bp[lid] = (const float*)d_in[din + 1];
      cinL[lid] = fpDims[i][l];
      coutL[lid] = fpDims[i][l + 1];
    }
  for (int lid = 0; lid < 32; ++lid) {
    int KP = P32h(cinL[lid]), NP = P32h(coutL[lid]);
    WtL[lid] = (_Float16*)walloc((size_t)KP * NP * 2);
    pack_w_kernel<<<cdivh((long long)KP * NP, 256), 256, 0, stream>>>(
        Wp[lid], WtL[lid], cinL[lid], coutL[lid], KP, NP);
  }

  // ---- split point cloud ----
  split_pc_kernel<<<cdivh(4 * 8192, 256), 256, 0, stream>>>(pc, xyzL[0], featL[0]);

  int curC[5] = {6, 0, 0, 0, 0};

  // =========================== SA stages ===========================
  for (int st = 0; st < 4; ++st) {
    int Ns = Nl[st], M = Nl[st + 1];
    fps_kernel<<<4, 1024, 0, stream>>>(xyzL[st], Ns, M, fpsIdx);
    gather_xyz_kernel<<<cdivh(4 * M, 256), 256, 0, stream>>>(xyzL[st], fpsIdx,
                                                             xyzL[st + 1], M, Ns);
    int totC = saDims[st][0][3] + saDims[st][1][3];
    int colofs = 0;
    for (int sc = 0; sc < 2; ++sc) {
      int K = saK[sc];
      float r = saR[st][sc];
      int rows = 4 * M * K, rowsP = rows;            // always multiple of 16
      ball_query_kernel<<<cdivh((long long)4 * M * 32, 256), 256, 0, stream>>>(
          xyzL[st + 1], xyzL[st], M, Ns, r * r, K, ballIdx);
      int C0 = saDims[st][sc][0];
      int P0 = P32h(C0);
      build_sa_x_kernel<<<cdivh((long long)rowsP * P0, 256), 256, 0, stream>>>(
          xyzL[st], featL[st], xyzL[st + 1], ballIdx, Xb, M, Ns, K, curC[st],
          P0, rows, rowsP);
      const _Float16* in = Xb;
      int KPc = P0;
      for (int l = 0; l < 3; ++l) {
        int lid = (st * 2 + sc) * 3 + l;
        int co = saDims[st][sc][l + 1];
        int NP = P32h(co);
        if (l < 2) {
          _Float16* out = (l == 0) ? Y0 : Y1;
          int total = (NP / 16) * (rowsP / 16);
          gemm_mlp_kernel<1, 0><<<cdivh(total, 8), 256, 0, stream>>>(
              in, WtL[lid], Bp[lid], out, nullptr, nullptr,
              rows, rowsP, KPc, NP, co, 0, 0, 0);
          in = out; KPc = NP;
        } else if (K == 16) {                        // fused max-pool, 1 tile
          int total = (NP / 16) * (rowsP / 16);
          gemm_mlp_kernel<1, 3><<<cdivh(total, 8), 256, 0, stream>>>(
              in, WtL[lid], Bp[lid], nullptr, nullptr, featL[st + 1],
              rows, rowsP, KPc, NP, co, 0, totC, colofs);
          colofs += co;
        } else {                                     // fused max-pool, 2 tiles
          int total = (NP / 16) * ((rowsP / 16) / 2);
          gemm_mlp_kernel<2, 3><<<cdivh(total, 8), 256, 0, stream>>>(
              in, WtL[lid], Bp[lid], nullptr, nullptr, featL[st + 1],
              rows, rowsP, KPc, NP, co, 0, totC, colofs);
          colofs += co;
        }
      }
    }
    curC[st + 1] = totC;   // 96, 256, 512, 1024
  }

  // =========================== FP stages ===========================
  for (int i = 3; i >= 0; --i) {
    int Nu = Nl[i], Nk = Nl[i + 1];
    int Ck = curC[i + 1], Cu = curC[i];
    three_nn_kernel<<<cdivh(4 * Nu, 256), 256, 0, stream>>>(
        xyzL[i], xyzL[i + 1], Nu, Nk, nnIdx, nnW);
    int cin0 = Ck + Cu;                   // == fpDims[i][0]
    int P0 = P32h(cin0);
    int rows = 4 * Nu;
    int rowsP = (rows + 15) & ~15;
    build_fp_x_kernel<<<cdivh((long long)rowsP * P0, 256), 256, 0, stream>>>(
        featL[i + 1], featL[i], nnIdx, nnW, Xb, Nu, Nk, Ck, curC[i + 1],
        Cu, curC[i], P0, rows, rowsP);
    // layer 0
    int lid0 = 24 + i * 2;
    int co1 = fpDims[i][1], NP1 = P32h(co1);
    {
      int total = (NP1 / 16) * (rowsP / 16);
      gemm_mlp_kernel<1, 0><<<cdivh(total, 8), 256, 0, stream>>>(
          Xb, WtL[lid0], Bp[lid0], Y0, nullptr, nullptr,
          rows, rowsP, P0, NP1, co1, 0, 0, 0);
    }
    // layer 1 (final, f32 out)
    int lid1 = 24 + i * 2 + 1;
    int co2 = fpDims[i][2], NP2 = P32h(co2);
    int total = (NP2 / 16) * (rowsP / 16);
    if (i > 0) {
      gemm_mlp_kernel<1, 1><<<cdivh(total, 8), 256, 0, stream>>>(
          Y0, WtL[lid1], Bp[lid1], nullptr, featL[i], nullptr,
          rows, rowsP, NP1, NP2, co2, co2, 0, 0);
      curC[i] = co2;
    } else {
      // write straight into d_out transposed: [B, 128, 8192]
      gemm_mlp_kernel<1, 2><<<cdivh(total, 8), 256, 0, stream>>>(
          Y0, WtL[lid1], Bp[lid1], nullptr, (float*)d_out, nullptr,
          rows, rowsP, NP1, NP2, co2, 0, 0, 0);
    }
  }
  (void)Y1;
}
